// MHA_83820581749265
// MI455X (gfx1250) — compile-verified, matrix operand order checked by reading, not run on hardware
//
#include <hip/hip_runtime.h>

// ---------------------------------------------------------------------------
// MHA with KV cache for MI455X (gfx1250), bf16 WMMA pipeline, f32 accumulate.
// B=2, T=2048, D_MODEL=2048, H=16, D_HEAD=128, CACHE=2048, KV=4096.
// GEMMs: A panel staged block-wide in LDS via GLOBAL_LOAD_ASYNC_TO_LDS_B128,
//        each wave computes a 16x64 tile (4 WMMA accumulators, A reused 4x).
// ---------------------------------------------------------------------------

#define B_     2
#define T_     2048
#define DM_    2048
#define H_     16
#define DH_    128
#define CACHE_ 2048
#define KV_    4096
#define QKSCALE 0.08838834764831845f  // 1/sqrt(128)

typedef __attribute__((ext_vector_type(16))) __bf16          v16bf;
typedef __attribute__((ext_vector_type(8)))  float           v8f;
typedef __attribute__((ext_vector_type(8)))  unsigned short  v8us;

union Frag { v16bf v; v8us h[2]; };

static __device__ __forceinline__ unsigned short f2bf(float f) {
    unsigned u = __builtin_bit_cast(unsigned, f);
    unsigned r = u + 0x7FFFu + ((u >> 16) & 1u);   // round-to-nearest-even
    return (unsigned short)(r >> 16);
}

static __device__ __forceinline__ v8f wmma_bf16(v16bf a, v16bf b, v8f c) {
    return __builtin_amdgcn_wmma_f32_16x16x32_bf16(
        /*neg_a=*/false, a, /*neg_b=*/false, b,
        /*c_mod=*/(short)0, c, /*reuse_a=*/false, /*reuse_b=*/false);
}

// A-fragment loader (16x32 bf16, row-major source, contiguous along K).
// Lane L: row = row0 + (L&15), half = L>>4.
// elems 0..7  -> K = k0 + half*8 + e ; elems 8..15 -> K = k0 + 16 + half*8 + e
// Doubles as B loader when the source is "N-major" (W[out,in], K[kv,d], Vt[d,kv]).
static __device__ __forceinline__ v16bf load_frag(const unsigned short* base,
                                                  int ld, int row0, int k0,
                                                  int lane) {
    const int m    = row0 + (lane & 15);
    const int half = (lane >> 4) & 1;
    const unsigned short* p = base + (size_t)m * ld + k0 + half * 8;
    Frag f;
    f.h[0] = *(const v8us*)(p);
    f.h[1] = *(const v8us*)(p + 16);
    return f.v;
}

// Async 16B copy global -> LDS (no VGPR round trip, tracked by ASYNCcnt).
static __device__ __forceinline__ void async_copy_b128(unsigned lds_byte_addr,
                                                       const void* gaddr) {
    unsigned long long ga = (unsigned long long)(uintptr_t)gaddr;
    __asm__ volatile("global_load_async_to_lds_b128 %0, %1, off"
                     :: "v"(lds_byte_addr), "v"(ga) : "memory");
}
static __device__ __forceinline__ void async_wait0() {
    __asm__ volatile("s_wait_asynccnt 0x0" ::: "memory");
}

// ---------------------------------------------------------------------------
// Elementwise f32 -> bf16
// ---------------------------------------------------------------------------
__global__ void cvt_f32_bf16(const float* __restrict__ in,
                             unsigned short* __restrict__ out, size_t n) {
    size_t i = (size_t)blockIdx.x * blockDim.x + threadIdx.x;
    size_t stride = (size_t)gridDim.x * blockDim.x;
    for (; i < n; i += stride) out[i] = f2bf(in[i]);
}

// cache_k -> kbf[b,h,0..CACHE,d] ; cache_v -> vtbf[b,h,d,0..CACHE] (transposed)
__global__ void cvt_cache(const float* __restrict__ ck,
                          const float* __restrict__ cv,
                          unsigned short* __restrict__ kbf,
                          unsigned short* __restrict__ vtbf) {
    const size_t total = (size_t)B_ * H_ * CACHE_ * DH_;
    size_t i = (size_t)blockIdx.x * blockDim.x + threadIdx.x;
    size_t stride = (size_t)gridDim.x * blockDim.x;
    for (; i < total; i += stride) {
        size_t d  = i & (DH_ - 1);
        size_t s  = (i >> 7) & (CACHE_ - 1);
        size_t bh = i >> 18;                       // b*H_+h
        kbf [(bh * KV_ + s) * DH_ + d]  = f2bf(ck[i]);
        vtbf[(bh * DH_ + d) * KV_ + s]  = f2bf(cv[i]);
    }
}

// ---------------------------------------------------------------------------
// Shared GEMM core: block = 256 thr (8 waves).
// A panel [16 x 2048] staged in LDS via async copies; wave w computes cols
// [col0, col0+64) with 4 accumulators; returns accumulators in acc[4].
// ---------------------------------------------------------------------------
static __device__ __forceinline__ void gemm_tile_16x64(
    const unsigned short* __restrict__ abf,   // [rows, DM_] row-major
    const unsigned short* __restrict__ wbf,   // [cols, DM_] row-major (B^T)
    unsigned short* lA,                       // LDS, 16*DM_ elems
    int row0, int col0, int lane, v8f acc[4]) {
    // ---- stage A panel: 16 rows x 2048 cols bf16 = 64KB = 4096 x 16B chunks
    const int tid = threadIdx.x;
    for (int c = tid; c < 16 * DM_ / 8; c += 256) {
        int row  = c >> 7;              // 128 x 16B chunks per row
        int col8 = (c & 127) * 8;
        unsigned lds_off = (unsigned)(uintptr_t)(&lA[c * 8]);
        async_copy_b128(lds_off, abf + (size_t)(row0 + row) * DM_ + col8);
    }
    async_wait0();
    __syncthreads();

    acc[0] = (v8f){}; acc[1] = (v8f){}; acc[2] = (v8f){}; acc[3] = (v8f){};
    for (int k0 = 0; k0 < DM_; k0 += 32) {
        v16bf a = load_frag(lA, DM_, 0, k0, lane);   // ds_load_b128 x2
#pragma unroll
        for (int j = 0; j < 4; ++j) {
            v16bf b = load_frag(wbf, DM_, col0 + j * 16, k0, lane);
            acc[j] = wmma_bf16(a, b, acc[j]);
        }
    }
}

// ---------------------------------------------------------------------------
// QKV projection: qkv = x @ Wqkv^T, scattered into Q / K(cache tail) / Vt.
// Grid: (6144/512, 4096/16); block 256 thr; wave -> 16x64 tile.
// ---------------------------------------------------------------------------
__global__ __launch_bounds__(256) void qkv_gemm(
    const unsigned short* __restrict__ xbf,    // [B*T, DM]
    const unsigned short* __restrict__ wbf,    // [3*DM, DM]
    unsigned short* __restrict__ qbf,          // [B,H,T,DH]
    unsigned short* __restrict__ kbf,          // [B,H,KV,DH]
    unsigned short* __restrict__ vtbf) {       // [B,H,DH,KV]
    __shared__ __align__(16) unsigned short lA[16 * DM_];
    const int lane = threadIdx.x & 31;
    const int wave = threadIdx.x >> 5;
    const int row0 = blockIdx.y * 16;
    const int col0 = blockIdx.x * 512 + wave * 64;

    v8f acc[4];
    gemm_tile_16x64(xbf, wbf, lA, row0, col0, lane, acc);

    const int half = lane >> 4, n = lane & 15;
#pragma unroll
    for (int j = 0; j < 4; ++j) {
        const int f = col0 + j * 16 + n;
#pragma unroll
        for (int r = 0; r < 8; ++r) {
            int row = row0 + r + 8 * half;
            int b = row >> 11, t = row & (T_ - 1);
            float v = acc[j][r];
            if (f < DM_) {                                    // Q (pre-scaled)
                int h = f >> 7, d = f & (DH_ - 1);
                qbf[(((size_t)b * H_ + h) * T_ + t) * DH_ + d] =
                    f2bf(v * QKSCALE);
            } else if (f < 2 * DM_) {                         // K at cache tail
                int g = f - DM_; int h = g >> 7, d = g & (DH_ - 1);
                kbf[(((size_t)b * H_ + h) * KV_ + (CACHE_ + t)) * DH_ + d] =
                    f2bf(v);
            } else {                                          // V transposed
                int g = f - 2 * DM_; int h = g >> 7, d = g & (DH_ - 1);
                vtbf[(((size_t)b * H_ + h) * DH_ + d) * KV_ + (CACHE_ + t)] =
                    f2bf(v);
            }
        }
    }
}

// ---------------------------------------------------------------------------
// Flash attention: block 128 thr = 4 waves, each wave owns 16 query rows.
// kv tiles of 32: 8 score WMMAs + 8 p@v WMMAs per tile, online softmax,
// P transposed C-layout -> A-layout through per-wave LDS tile.
// ---------------------------------------------------------------------------
__global__ __launch_bounds__(128) void attn_flash(
    const unsigned short* __restrict__ qbf,   // [B,H,T,DH]
    const unsigned short* __restrict__ kbf,   // [B,H,KV,DH]
    const unsigned short* __restrict__ vtbf,  // [B,H,DH,KV]
    unsigned short* __restrict__ ybf) {       // [B,T,DM]
    __shared__ __align__(16) unsigned short ldsP[4][16 * 32];

    const int lane = threadIdx.x & 31;
    const int wave = threadIdx.x >> 5;
    const int b = blockIdx.z, h = blockIdx.y;
    const int q0 = blockIdx.x * 64 + wave * 16;

    const unsigned short* qp = qbf  + ((size_t)b * H_ + h) * (size_t)T_  * DH_;
    const unsigned short* kp = kbf  + ((size_t)b * H_ + h) * (size_t)KV_ * DH_;
    const unsigned short* vp = vtbf + ((size_t)b * H_ + h) * (size_t)DH_ * KV_;

    v16bf qf[4];
#pragma unroll
    for (int c = 0; c < 4; ++c) qf[c] = load_frag(qp, DH_, q0, c * 32, lane);

    v8f acc[8];
    float m[8], l[8];
#pragma unroll
    for (int r = 0; r < 8; ++r) { acc[r] = (v8f){}; m[r] = -1e30f; l[r] = 0.f; }

    const int half = lane >> 4, n = lane & 15;
    unsigned short* pl = ldsP[wave];
    const int kv_end = ((CACHE_ + q0 + 16) + 31) & ~31;   // causal frontier

    for (int kv = 0; kv < kv_end; kv += 32) {
        v8f s0 = {}, s1 = {};
#pragma unroll
        for (int c = 0; c < 4; ++c) {
            v16bf kb0 = load_frag(kp, DH_, kv,      c * 32, lane);
            s0 = wmma_bf16(qf[c], kb0, s0);
            v16bf kb1 = load_frag(kp, DH_, kv + 16, c * 32, lane);
            s1 = wmma_bf16(qf[c], kb1, s1);
        }
        if (kv + 31 > CACHE_ + q0) {                      // diagonal tile: mask
#pragma unroll
            for (int r = 0; r < 8; ++r) {
                int qpos = CACHE_ + q0 + r + 8 * half;
                if (kv + n      > qpos) s0[r] = -1e30f;
                if (kv + 16 + n > qpos) s1[r] = -1e30f;
            }
        }
        float p0[8], p1[8];
#pragma unroll
        for (int r = 0; r < 8; ++r) {
            float rm = fmaxf(s0[r], s1[r]);
            rm = fmaxf(rm, __shfl_xor(rm, 1, 32));
            rm = fmaxf(rm, __shfl_xor(rm, 2, 32));
            rm = fmaxf(rm, __shfl_xor(rm, 4, 32));
            rm = fmaxf(rm, __shfl_xor(rm, 8, 32));
            float mn = fmaxf(m[r], rm);
            float alpha = __expf(m[r] - mn);
            m[r] = mn;
            p0[r] = __expf(s0[r] - mn);
            p1[r] = __expf(s1[r] - mn);
            float rs = p0[r] + p1[r];
            rs += __shfl_xor(rs, 1, 32);
            rs += __shfl_xor(rs, 2, 32);
            rs += __shfl_xor(rs, 4, 32);
            rs += __shfl_xor(rs, 8, 32);
            l[r] = l[r] * alpha + rs;
#pragma unroll
            for (int dt = 0; dt < 8; ++dt) acc[dt][r] *= alpha;
        }
        // C-layout P -> LDS (16x32 row-major), reload as A-fragment.
#pragma unroll
        for (int r = 0; r < 8; ++r) {
            int row = r + 8 * half;
            pl[row * 32 + n]      = f2bf(p0[r]);
            pl[row * 32 + 16 + n] = f2bf(p1[r]);
        }
        __asm__ volatile("s_wait_dscnt 0x0" ::: "memory"); // in-wave RAW on LDS
        v16bf pf = load_frag(pl, 32, 0, 0, lane);
#pragma unroll
        for (int dt = 0; dt < 8; ++dt) {
            v16bf vf = load_frag(vp, KV_, dt * 16, kv, lane);
            acc[dt] = wmma_bf16(pf, vf, acc[dt]);
        }
    }

#pragma unroll
    for (int r = 0; r < 8; ++r) {
        float inv = 1.0f / l[r];
        int t = q0 + r + 8 * half;
        size_t o = ((size_t)b * T_ + t) * DM_ + (size_t)h * DH_;
#pragma unroll
        for (int dt = 0; dt < 8; ++dt)
            ybf[o + dt * 16 + n] = f2bf(acc[dt][r] * inv);
    }
}

// ---------------------------------------------------------------------------
// Output projection: out = y @ Wproj^T, f32 result straight to d_out.
// Grid: (2048/512, 4096/16); block 256 thr; wave -> 16x64 tile.
// ---------------------------------------------------------------------------
__global__ __launch_bounds__(256) void proj_gemm(
    const unsigned short* __restrict__ ybf,   // [B*T, DM]
    const unsigned short* __restrict__ wbf,   // [DM, DM]
    float* __restrict__ out) {                // [B*T, DM]
    __shared__ __align__(16) unsigned short lA[16 * DM_];
    const int lane = threadIdx.x & 31;
    const int wave = threadIdx.x >> 5;
    const int row0 = blockIdx.y * 16;
    const int col0 = blockIdx.x * 512 + wave * 64;

    v8f acc[4];
    gemm_tile_16x64(ybf, wbf, lA, row0, col0, lane, acc);

    const int half = lane >> 4, n = lane & 15;
#pragma unroll
    for (int j = 0; j < 4; ++j)
#pragma unroll
        for (int r = 0; r < 8; ++r)
            out[(size_t)(row0 + r + 8 * half) * DM_ + col0 + j * 16 + n] =
                acc[j][r];
}

// ---------------------------------------------------------------------------
extern "C" void kernel_launch(void* const* d_in, const int* in_sizes, int n_in,
                              void* d_out, int out_size, void* d_ws,
                              size_t ws_size, hipStream_t stream) {
    const float* x     = (const float*)d_in[0];
    const float* Wqkv  = (const float*)d_in[1];
    const float* Wproj = (const float*)d_in[2];
    const float* ck    = (const float*)d_in[3];
    const float* cv    = (const float*)d_in[4];
    float* out = (float*)d_out;

    unsigned short* ws = (unsigned short*)d_ws;
    size_t off = 0;
    unsigned short* xbf     = ws + off; off += (size_t)B_ * T_ * DM_;       // 8M
    unsigned short* wqkvbf  = ws + off; off += (size_t)3 * DM_ * DM_;       // 12M
    unsigned short* wprojbf = ws + off; off += (size_t)DM_ * DM_;           // 4M
    unsigned short* qbf     = ws + off; off += (size_t)B_ * H_ * T_ * DH_;  // 8M
    unsigned short* kbf     = ws + off; off += (size_t)B_ * H_ * KV_ * DH_; // 16M
    unsigned short* vtbf    = ws + off; off += (size_t)B_ * H_ * DH_ * KV_; // 16M
    unsigned short* ybf     = ws + off; off += (size_t)B_ * T_ * DM_;       // 8M

    cvt_f32_bf16<<<2048, 256, 0, stream>>>(x, xbf, (size_t)B_ * T_ * DM_);
    cvt_f32_bf16<<<2048, 256, 0, stream>>>(Wqkv, wqkvbf, (size_t)3 * DM_ * DM_);
    cvt_f32_bf16<<<2048, 256, 0, stream>>>(Wproj, wprojbf, (size_t)DM_ * DM_);
    cvt_cache<<<2048, 256, 0, stream>>>(ck, cv, kbf, vtbf);

    qkv_gemm<<<dim3(3 * DM_ / 512, B_ * T_ / 16), 256, 0, stream>>>(
        xbf, wqkvbf, qbf, kbf, vtbf);

    attn_flash<<<dim3(T_ / 64, H_, B_), 128, 0, stream>>>(qbf, kbf, vtbf, ybf);

    proj_gemm<<<dim3(DM_ / 512, B_ * T_ / 16), 256, 0, stream>>>(
        ybf, wprojbf, out);
}
